// PreLNTransformerBlock_25065429139810
// MI455X (gfx1250) — matrix operand order repaired and checked by reading, not compiled, not run on hardware
//
#include <hip/hip_runtime.h>

// Pre-LN transformer block for MI455X (gfx1250, wave32).
// GEMMs run on v_wmma_f32_16x16x32_bf16 with split-precision bf16 (hi+lo)
// operands: A*B ~= Ah*Bh + Al*Bh + Ah*Bl, f32 accumulate (~2^-16 rel err).
// All reused operands are pre-packed to bf16 hi/lo in memory-bound producer
// epilogues so the GEMM inner loops are pure global_load_b128 + v_wmma.
// Only attn-probs (must exist as fp32 in d_out) are split on the fly, using
// cheap v_perm_b32 truncating packs.

#define B_   2
#define T_   2048
#define C_   768
#define H_   12
#define D_   64
#define HID_ 3072

typedef __attribute__((ext_vector_type(16))) __bf16 v16bf;
typedef __attribute__((ext_vector_type(8)))  float  v8f;
typedef unsigned short ushort_t;

struct BfFrag { uint4 a, b; };   // 16 bf16 values, element i = ushort i
struct F8     { float f[8]; };

// ---------- scalar bf16 split helpers (epilogues / pack kernels) ----------
__device__ __forceinline__ ushort_t bf_rne(float f) {
  unsigned u = __float_as_uint(f);
  u += 0x7FFFu + ((u >> 16) & 1u);
  return (ushort_t)(u >> 16);
}
__device__ __forceinline__ void store_split(float v, ushort_t* __restrict__ hi,
                                            ushort_t* __restrict__ lo, size_t idx) {
  unsigned uh = __float_as_uint(v) & 0xffff0000u;       // hi = truncate
  hi[idx] = (ushort_t)(uh >> 16);
  lo[idx] = bf_rne(v - __uint_as_float(uh));            // lo = RNE(residual)
}

// ---------- packed-bf16 fragment loads ------------------------------------
// A fragment (16x32, MxK): lane r=lane%16 -> M; K chunks of 8:
//   elem i<8: K=8g+i ; elem i>=8: K=8g+8+i      (g = lane/16)
__device__ __forceinline__ void load_a_pk(const ushort_t* __restrict__ base, int ld,
                                          int row0, int col0, BfFrag& out) {
  const int lane = threadIdx.x & 31;
  const int r = lane & 15, g = lane >> 4;
  const ushort_t* p = base + (size_t)(row0 + r) * ld + col0 + 8 * g;
  out.a = *(const uint4*)p;          // K = 8g .. 8g+7
  out.b = *(const uint4*)(p + 16);   // K = 8g+16 .. 8g+23
}
// B fragment (32x16, KxN) for A @ W^T, W row-major [N,K]: lane r -> N;
//   elem i<8: K=8g+i ; elem i>=8: K=8g+8+i      (g = lane/16)
__device__ __forceinline__ void load_b_pk(const ushort_t* __restrict__ base, int ld,
                                          int row0, int col0, BfFrag& out) {
  const int lane = threadIdx.x & 31;
  const int r = lane & 15, g = lane >> 4;
  const ushort_t* p = base + (size_t)(row0 + r) * ld + col0 + 8 * g;
  out.a = *(const uint4*)p;
  out.b = *(const uint4*)(p + 16);
}

// ---------- on-the-fly fp32 A fragment with v_perm truncating split -------
__device__ __forceinline__ unsigned pack_hi_pair(float a, float b) {
  // result = [bf16(b) : bf16(a)] (truncation), one v_perm_b32
  return __builtin_amdgcn_perm(__float_as_uint(b), __float_as_uint(a), 0x07060302u);
}
__device__ __forceinline__ void load_a_f32_split(const float* __restrict__ base, int ld,
                                                 int row0, int col0,
                                                 BfFrag& hi, BfFrag& lo) {
  const int lane = threadIdx.x & 31;
  const int r = lane & 15, g = lane >> 4;
  const float* p = base + (size_t)(row0 + r) * ld + col0 + 8 * g;
  float4 f0 = ((const float4*)p)[0], f1 = ((const float4*)p)[1];
  float4 f2 = ((const float4*)(p + 16))[0], f3 = ((const float4*)(p + 16))[1];
  float v[16];
  v[0]=f0.x;  v[1]=f0.y;  v[2]=f0.z;  v[3]=f0.w;
  v[4]=f1.x;  v[5]=f1.y;  v[6]=f1.z;  v[7]=f1.w;
  v[8]=f2.x;  v[9]=f2.y;  v[10]=f2.z; v[11]=f2.w;
  v[12]=f3.x; v[13]=f3.y; v[14]=f3.z; v[15]=f3.w;
  unsigned wh[8], wl[8];
#pragma unroll
  for (int i = 0; i < 8; ++i) {
    float a = v[2 * i], b = v[2 * i + 1];
    wh[i] = pack_hi_pair(a, b);
    float la = a - __uint_as_float(__float_as_uint(a) & 0xffff0000u);
    float lb = b - __uint_as_float(__float_as_uint(b) & 0xffff0000u);
    wl[i] = pack_hi_pair(la, lb);
  }
  hi.a.x = wh[0]; hi.a.y = wh[1]; hi.a.z = wh[2]; hi.a.w = wh[3];
  hi.b.x = wh[4]; hi.b.y = wh[5]; hi.b.z = wh[6]; hi.b.w = wh[7];
  lo.a.x = wl[0]; lo.a.y = wl[1]; lo.a.z = wl[2]; lo.a.w = wl[3];
  lo.b.x = wl[4]; lo.b.y = wl[5]; lo.b.z = wl[6]; lo.b.w = wl[7];
}

// ---------- split-precision MMA -------------------------------------------
__device__ __forceinline__ void mma3(v8f& acc, const BfFrag& ah, const BfFrag& al,
                                     const BfFrag& bh, const BfFrag& bl) {
  v16bf AH = __builtin_bit_cast(v16bf, ah);
  v16bf AL = __builtin_bit_cast(v16bf, al);
  v16bf BH = __builtin_bit_cast(v16bf, bh);
  v16bf BL = __builtin_bit_cast(v16bf, bl);
  acc = __builtin_amdgcn_wmma_f32_16x16x32_bf16(false, AH, false, BH, (short)0, acc, false, false);
  acc = __builtin_amdgcn_wmma_f32_16x16x32_bf16(false, AL, false, BH, (short)0, acc, false, false);
  acc = __builtin_amdgcn_wmma_f32_16x16x32_bf16(false, AH, false, BL, (short)0, acc, false, false);
}

// ---------- 32x32 wave tile, both operands pre-packed ---------------------
__device__ __forceinline__ void tile_pk(const ushort_t* __restrict__ Ah,
                                        const ushort_t* __restrict__ Al, int lda,
                                        const ushort_t* __restrict__ Wh,
                                        const ushort_t* __restrict__ Wl, int ldw,
                                        int m0, int n0, int K, v8f acc[2][2]) {
  const int r = threadIdx.x & 15;
  for (int k = 0; k < K; k += 32) {
    if (k + 32 < K) {  // global_prefetch_b8
      __builtin_prefetch(Ah + (size_t)(m0 + r) * lda + k + 32, 0, 3);
      __builtin_prefetch(Wh + (size_t)(n0 + r) * ldw + k + 32, 0, 3);
    }
    BfFrag ah[2], al[2], bh[2], bl[2];
    load_a_pk(Ah, lda, m0,      k, ah[0]);
    load_a_pk(Ah, lda, m0 + 16, k, ah[1]);
    load_a_pk(Al, lda, m0,      k, al[0]);
    load_a_pk(Al, lda, m0 + 16, k, al[1]);
    load_b_pk(Wh, ldw, n0,      k, bh[0]);
    load_b_pk(Wh, ldw, n0 + 16, k, bh[1]);
    load_b_pk(Wl, ldw, n0,      k, bl[0]);
    load_b_pk(Wl, ldw, n0 + 16, k, bl[1]);
#pragma unroll
    for (int mi = 0; mi < 2; ++mi)
#pragma unroll
      for (int ni = 0; ni < 2; ++ni)
        mma3(acc[mi][ni], ah[mi], al[mi], bh[ni], bl[ni]);
  }
}

// ---------- 32x32 wave tile, A = fp32 (attn probs), W packed --------------
__device__ __forceinline__ void tile_av(const float* __restrict__ A, int lda,
                                        const ushort_t* __restrict__ Wh,
                                        const ushort_t* __restrict__ Wl, int ldw,
                                        int m0, int n0, int K, v8f acc[2][2]) {
  for (int k = 0; k < K; k += 32) {
    BfFrag ah[2], al[2], bh[2], bl[2];
    load_a_f32_split(A, lda, m0,      k, ah[0], al[0]);
    load_a_f32_split(A, lda, m0 + 16, k, ah[1], al[1]);
    load_b_pk(Wh, ldw, n0,      k, bh[0]);
    load_b_pk(Wh, ldw, n0 + 16, k, bh[1]);
    load_b_pk(Wl, ldw, n0,      k, bl[0]);
    load_b_pk(Wl, ldw, n0 + 16, k, bl[1]);
#pragma unroll
    for (int mi = 0; mi < 2; ++mi)
#pragma unroll
      for (int ni = 0; ni < 2; ++ni)
        mma3(acc[mi][ni], ah[mi], al[mi], bh[ni], bl[ni]);
  }
}

// ---------- reductions ----------------------------------------------------
__device__ __forceinline__ float wave_max(float v) {
#pragma unroll
  for (int off = 16; off > 0; off >>= 1) v = fmaxf(v, __shfl_xor(v, off, 32));
  return v;
}
__device__ __forceinline__ float wave_sum(float v) {
#pragma unroll
  for (int off = 16; off > 0; off >>= 1) v += __shfl_xor(v, off, 32);
  return v;
}

// ---------- weight pack: fp32 -> bf16 hi/lo -------------------------------
__global__ void pack_kernel(const float* __restrict__ src, ushort_t* __restrict__ hi,
                            ushort_t* __restrict__ lo, int n) {
  int idx = blockIdx.x * 256 + threadIdx.x;
  if (idx < n) store_split(src[idx], hi, lo, idx);
}

// ---------- LayerNorm -> packed bf16 hi/lo (C=768=3*256) ------------------
__global__ void ln_kernel(const float* __restrict__ x, const float* __restrict__ g,
                          const float* __restrict__ b,
                          ushort_t* __restrict__ ohi, ushort_t* __restrict__ olo) {
  __shared__ float sm[8];
  const size_t row = blockIdx.x;
  const float* px = x + row * C_;
  const int t = threadIdx.x;
  float v0 = px[t], v1 = px[t + 256], v2 = px[t + 512];
  float s = wave_sum(v0 + v1 + v2);
  if ((t & 31) == 0) sm[t >> 5] = s;
  __syncthreads();
  float tot = 0.f;
#pragma unroll
  for (int i = 0; i < 8; ++i) tot += sm[i];
  __syncthreads();
  const float mu = tot * (1.0f / C_);
  float d0 = v0 - mu, d1 = v1 - mu, d2 = v2 - mu;
  s = wave_sum(d0 * d0 + d1 * d1 + d2 * d2);
  if ((t & 31) == 0) sm[t >> 5] = s;
  __syncthreads();
  tot = 0.f;
#pragma unroll
  for (int i = 0; i < 8; ++i) tot += sm[i];
  const float inv = rsqrtf(tot * (1.0f / C_) + 1e-5f);
  const size_t o = row * C_;
  store_split(d0 * inv * g[t]       + b[t],       ohi, olo, o + t);
  store_split(d1 * inv * g[t + 256] + b[t + 256], ohi, olo, o + t + 256);
  store_split(d2 * inv * g[t + 512] + b[t + 512], ohi, olo, o + t + 512);
}

// ---------- QKV GEMM: epilogue scatters packed q/k [B,H,T,D], vT [B,H,D,T]
__global__ void gemm_qkv_kernel(const ushort_t* __restrict__ Ah, const ushort_t* __restrict__ Al,
                                const ushort_t* __restrict__ Wh, const ushort_t* __restrict__ Wl,
                                const float* __restrict__ bias,
                                ushort_t* __restrict__ qh, ushort_t* __restrict__ ql,
                                ushort_t* __restrict__ kh, ushort_t* __restrict__ kl,
                                ushort_t* __restrict__ vth, ushort_t* __restrict__ vtl) {
  const int wave = threadIdx.x >> 5;
  const int m0 = blockIdx.x * 128 + (wave & 3) * 32;
  const int n0 = blockIdx.y * 64 + (wave >> 2) * 32;
  v8f acc[2][2] = {};
  tile_pk(Ah, Al, C_, Wh, Wl, C_, m0, n0, C_, acc);
  const int lane = threadIdx.x & 31, r = lane & 15, g = lane >> 4;
#pragma unroll
  for (int mi = 0; mi < 2; ++mi)
#pragma unroll
    for (int ni = 0; ni < 2; ++ni) {
      F8 a = __builtin_bit_cast(F8, acc[mi][ni]);
      const int n = n0 + ni * 16 + r;
      const int sec = (n0 + ni * 16) / C_;   // 0=q 1=k 2=v (uniform per frag)
      const int c = n - sec * C_;
      const int h = c >> 6, d = c & 63;
      const float bn = bias[n];
#pragma unroll
      for (int j = 0; j < 8; ++j) {
        const int m = m0 + mi * 16 + 8 * g + j;
        const int bb = m >> 11, t = m & (T_ - 1);
        const float v = a.f[j] + bn;
        const size_t bh = (size_t)bb * H_ + h;
        if (sec == 0)      store_split(v, qh, ql, (bh * T_ + t) * D_ + d);
        else if (sec == 1) store_split(v, kh, kl, (bh * T_ + t) * D_ + d);
        else               store_split(v, vth, vtl, (bh * D_ + d) * T_ + t);
      }
    }
}

// ---------- attention logits = q@k^T*scale + bias + mask (fp32 to d_out) --
__global__ void attn_logits_kernel(const ushort_t* __restrict__ qh, const ushort_t* __restrict__ ql,
                                   const ushort_t* __restrict__ kh, const ushort_t* __restrict__ kl,
                                   const float* __restrict__ bias, const float* __restrict__ mask,
                                   float* __restrict__ attn) {
  const int bh = blockIdx.z, b = bh / H_;
  const size_t hoff = (size_t)bh * T_ * D_;
  const int wave = threadIdx.x >> 5;
  const int m0 = blockIdx.x * 128 + (wave & 3) * 32;
  const int n0 = blockIdx.y * 64 + (wave >> 2) * 32;
  v8f acc[2][2] = {};
  tile_pk(qh + hoff, ql + hoff, D_, kh + hoff, kl + hoff, D_, m0, n0, D_, acc);
  const float* pb = bias + (size_t)bh * T_ * T_;
  const float* pm = mask + (size_t)b * T_ * T_;
  float* po = attn + (size_t)bh * T_ * T_;
  const float scale = 0.125f;
  const int lane = threadIdx.x & 31, r = lane & 15, g = lane >> 4;
#pragma unroll
  for (int mi = 0; mi < 2; ++mi)
#pragma unroll
    for (int ni = 0; ni < 2; ++ni) {
      F8 a = __builtin_bit_cast(F8, acc[mi][ni]);
      const int n = n0 + ni * 16 + r;
#pragma unroll
      for (int j = 0; j < 8; ++j) {
        const int m = m0 + mi * 16 + 8 * g + j;
        const size_t idx = (size_t)m * T_ + n;
        po[idx] = a.f[j] * scale + pb[idx] + pm[idx];
      }
    }
}

// ---------- softmax over last dim (T=2048), register-resident -------------
__global__ void softmax_kernel(float* __restrict__ attn) {
  __shared__ float sm[8];
  const size_t row = blockIdx.x;
  float* p = attn + row * (size_t)T_;
  const int t = threadIdx.x;
  float v[8];
#pragma unroll
  for (int i = 0; i < 8; ++i) v[i] = p[t + 256 * i];
  float m = v[0];
#pragma unroll
  for (int i = 1; i < 8; ++i) m = fmaxf(m, v[i]);
  m = wave_max(m);
  if ((t & 31) == 0) sm[t >> 5] = m;
  __syncthreads();
  float bm = sm[0];
#pragma unroll
  for (int i = 1; i < 8; ++i) bm = fmaxf(bm, sm[i]);
  __syncthreads();
  float s = 0.f;
#pragma unroll
  for (int i = 0; i < 8; ++i) { v[i] = expf(v[i] - bm); s += v[i]; }
  s = wave_sum(s);
  if ((t & 31) == 0) sm[t >> 5] = s;
  __syncthreads();
  float bs = 0.f;
#pragma unroll
  for (int i = 0; i < 8; ++i) bs += sm[i];
  const float inv = 1.0f / bs;
#pragma unroll
  for (int i = 0; i < 8; ++i) p[t + 256 * i] = v[i] * inv;
}

// ---------- y = attn @ v : A fp32 on-the-fly split, vT packed -------------
__global__ void attn_av_kernel(const float* __restrict__ attn,
                               const ushort_t* __restrict__ vth, const ushort_t* __restrict__ vtl,
                               ushort_t* __restrict__ yh, ushort_t* __restrict__ yl) {
  const int bh = blockIdx.z, b = bh / H_, h = bh - b * H_;
  const float* Ap = attn + (size_t)bh * T_ * T_;
  const size_t voff = (size_t)bh * D_ * T_;
  const int wave = threadIdx.x >> 5;
  const int m0 = blockIdx.x * 128 + (wave & 3) * 32;
  const int n0 = (wave >> 2) * 32;      // N = D = 64
  v8f acc[2][2] = {};
  tile_av(Ap, T_, vth + voff, vtl + voff, T_, m0, n0, T_, acc);
  const size_t ybase = (size_t)b * T_ * C_ + h * D_;
  const int lane = threadIdx.x & 31, r = lane & 15, g = lane >> 4;
#pragma unroll
  for (int mi = 0; mi < 2; ++mi)
#pragma unroll
    for (int ni = 0; ni < 2; ++ni) {
      F8 a = __builtin_bit_cast(F8, acc[mi][ni]);
      const int n = n0 + ni * 16 + r;
#pragma unroll
      for (int j = 0; j < 8; ++j) {
        const int m = m0 + mi * 16 + 8 * g + j;
        store_split(a.f[j], yh, yl, ybase + (size_t)m * C_ + n);
      }
    }
}

// ---------- generic packed GEMM: GELU->packed out, or layerscale residual -
enum { EPI_GELU_PACK = 0, EPI_RESID = 1 };

template <int EPI>
__global__ void gemm_pk_kernel(const ushort_t* __restrict__ Ah, const ushort_t* __restrict__ Al, int lda,
                               const ushort_t* __restrict__ Wh, const ushort_t* __restrict__ Wl, int ldw,
                               const float* __restrict__ bias,
                               const float* __restrict__ resid, const float* __restrict__ ls,
                               float* __restrict__ outF,
                               ushort_t* __restrict__ outHi, ushort_t* __restrict__ outLo,
                               int ldc, int K) {
  const int wave = threadIdx.x >> 5;
  const int m0 = blockIdx.x * 128 + (wave & 3) * 32;
  const int n0 = blockIdx.y * 64 + (wave >> 2) * 32;
  v8f acc[2][2] = {};
  tile_pk(Ah, Al, lda, Wh, Wl, ldw, m0, n0, K, acc);
  const int lane = threadIdx.x & 31, r = lane & 15, g = lane >> 4;
#pragma unroll
  for (int mi = 0; mi < 2; ++mi)
#pragma unroll
    for (int ni = 0; ni < 2; ++ni) {
      F8 a = __builtin_bit_cast(F8, acc[mi][ni]);
      const int n = n0 + ni * 16 + r;
      const float bn = bias[n];
      const float lsn = (EPI == EPI_RESID) ? ls[n] : 0.f;
#pragma unroll
      for (int j = 0; j < 8; ++j) {
        const int m = m0 + mi * 16 + 8 * g + j;
        float v = a.f[j] + bn;
        const size_t idx = (size_t)m * ldc + n;
        if (EPI == EPI_GELU_PACK) {
          v = 0.5f * v * (1.0f + erff(v * 0.70710678118654752440f));
          store_split(v, outHi, outLo, idx);
        } else {
          outF[idx] = resid[idx] + lsn * v;
        }
      }
    }
}

extern "C" void kernel_launch(void* const* d_in, const int* in_sizes, int n_in,
                              void* d_out, int out_size, void* d_ws, size_t ws_size,
                              hipStream_t stream) {
  const float* x      = (const float*)d_in[0];
  const float* mask   = (const float*)d_in[1];
  const float* bias   = (const float*)d_in[2];
  const float* ln1_g  = (const float*)d_in[3];
  const float* ln1_b  = (const float*)d_in[4];
  const float* qkv_w  = (const float*)d_in[5];
  const float* qkv_b  = (const float*)d_in[6];
  const float* proj_w = (const float*)d_in[7];
  const float* proj_b = (const float*)d_in[8];
  const float* ls1    = (const float*)d_in[9];
  const float* ln2_g  = (const float*)d_in[10];
  const float* ln2_b  = (const float*)d_in[11];
  const float* w1     = (const float*)d_in[12];
  const float* b1     = (const float*)d_in[13];
  const float* w2     = (const float*)d_in[14];
  const float* b2     = (const float*)d_in[15];
  const float* ls2    = (const float*)d_in[16];

  float* xout = (float*)d_out;                        // [B,T,C] fp32
  float* attn = xout + (size_t)B_ * T_ * C_;          // [B,H,T,T] fp32

  // ---- workspace carve (256B-aligned chunks) ----
  char* wp = (char*)d_ws;
  auto alloc = [&](size_t bytes) -> void* {
    void* p = wp; wp += (bytes + 255) & ~(size_t)255; return p;
  };
  const size_t BTC = (size_t)B_ * T_ * C_;            // 3,145,728
  ushort_t* xn_h = (ushort_t*)alloc(BTC * 2);
  ushort_t* xn_l = (ushort_t*)alloc(BTC * 2);
  ushort_t* qw_h = (ushort_t*)alloc((size_t)3 * C_ * C_ * 2);
  ushort_t* qw_l = (ushort_t*)alloc((size_t)3 * C_ * C_ * 2);
  ushort_t* pw_h = (ushort_t*)alloc((size_t)C_ * C_ * 2);
  ushort_t* pw_l = (ushort_t*)alloc((size_t)C_ * C_ * 2);
  ushort_t* w1_h = (ushort_t*)alloc((size_t)HID_ * C_ * 2);
  ushort_t* w1_l = (ushort_t*)alloc((size_t)HID_ * C_ * 2);
  ushort_t* w2_h = (ushort_t*)alloc((size_t)C_ * HID_ * 2);
  ushort_t* w2_l = (ushort_t*)alloc((size_t)C_ * HID_ * 2);
  ushort_t* q_h  = (ushort_t*)alloc(BTC * 2);
  ushort_t* q_l  = (ushort_t*)alloc(BTC * 2);
  ushort_t* k_h  = (ushort_t*)alloc(BTC * 2);
  ushort_t* k_l  = (ushort_t*)alloc(BTC * 2);
  ushort_t* vt_h = (ushort_t*)alloc(BTC * 2);
  ushort_t* vt_l = (ushort_t*)alloc(BTC * 2);
  ushort_t* y_h  = (ushort_t*)alloc(BTC * 2);
  ushort_t* y_l  = (ushort_t*)alloc(BTC * 2);
  float*    x1   = (float*)   alloc(BTC * 4);
  ushort_t* x1n_h= (ushort_t*)alloc(BTC * 2);
  ushort_t* x1n_l= (ushort_t*)alloc(BTC * 2);
  ushort_t* h1_h = (ushort_t*)alloc((size_t)B_ * T_ * HID_ * 2);
  ushort_t* h1_l = (ushort_t*)alloc((size_t)B_ * T_ * HID_ * 2);

  // ---- weight packs (tiny, memory-bound) ----
  pack_kernel<<<(3 * C_ * C_ + 255) / 256, 256, 0, stream>>>(qkv_w, qw_h, qw_l, 3 * C_ * C_);
  pack_kernel<<<(C_ * C_ + 255) / 256, 256, 0, stream>>>(proj_w, pw_h, pw_l, C_ * C_);
  pack_kernel<<<(HID_ * C_ + 255) / 256, 256, 0, stream>>>(w1, w1_h, w1_l, HID_ * C_);
  pack_kernel<<<(C_ * HID_ + 255) / 256, 256, 0, stream>>>(w2, w2_h, w2_l, C_ * HID_);

  // ---- block ----
  ln_kernel<<<B_ * T_, 256, 0, stream>>>(x, ln1_g, ln1_b, xn_h, xn_l);

  gemm_qkv_kernel<<<dim3((B_ * T_) / 128, (3 * C_) / 64), 256, 0, stream>>>(
      xn_h, xn_l, qw_h, qw_l, qkv_b, q_h, q_l, k_h, k_l, vt_h, vt_l);

  attn_logits_kernel<<<dim3(T_ / 128, T_ / 64, B_ * H_), 256, 0, stream>>>(
      q_h, q_l, k_h, k_l, bias, mask, attn);

  softmax_kernel<<<B_ * H_ * T_, 256, 0, stream>>>(attn);

  attn_av_kernel<<<dim3(T_ / 128, 1, B_ * H_), 256, 0, stream>>>(
      attn, vt_h, vt_l, y_h, y_l);

  gemm_pk_kernel<EPI_RESID><<<dim3((B_ * T_) / 128, C_ / 64), 256, 0, stream>>>(
      y_h, y_l, C_, pw_h, pw_l, C_, proj_b, x, ls1, x1, nullptr, nullptr, C_, C_);

  ln_kernel<<<B_ * T_, 256, 0, stream>>>(x1, ln2_g, ln2_b, x1n_h, x1n_l);

  gemm_pk_kernel<EPI_GELU_PACK><<<dim3((B_ * T_) / 128, HID_ / 64), 256, 0, stream>>>(
      x1n_h, x1n_l, C_, w1_h, w1_l, C_, b1, nullptr, nullptr, nullptr, h1_h, h1_l, HID_, C_);

  gemm_pk_kernel<EPI_RESID><<<dim3((B_ * T_) / 128, C_ / 64), 256, 0, stream>>>(
      h1_h, h1_l, HID_, w2_h, w2_l, HID_, b2, x1, ls2, xout, nullptr, nullptr, C_, HID_);
}
